// RNN_25125558682331
// MI455X (gfx1250) — matrix operand order repaired and checked by reading, not compiled, not run on hardware
//
#include <hip/hip_runtime.h>
#include <hip/hip_bf16.h>
#include <math.h>

// ---------------------------------------------------------------------------
// LSTM: N=64, T=512, F_IN=1024, H=1024.
//   Phase 1 (parallel): Xg[t*64+n][g] = x[n][t][:] @ Wx + bias   (bf16 WMMA)
//   Phase 2 (persistent): per step, A = h_{t-1} @ Wh + Xg[t]; LSTM gating.
// bf16 operands (v_wmma_f32_16x16x32_bf16), fp32 accumulation.
// Wh slice staged in LDS once via async global->LDS; B operand served from LDS.
// Ping-pong double-buffered fragments (no register rotation copies).
// ---------------------------------------------------------------------------

typedef __attribute__((ext_vector_type(16))) __bf16 v16bf;
typedef __attribute__((ext_vector_type(8)))  __bf16 v8bf;
typedef __attribute__((ext_vector_type(8)))  float  v8f;

#define NN   64
#define TT   512
#define FF   1024
#define HH   1024
#define G4   4096              // 4*H
#define NT   (NN*TT)           // 32768 rows of the big GEMM
#define RNN_BLOCKS 128         // persistent recurrent kernel grid
#define BS_BYTES   (256*1024)  // LDS: staged Wh fragments (8 waves x 32KB)
#define LDS_TOTAL  (BS_BYTES + 8*1024)

// ---------------- prep kernels ----------------------------------------------

// weight (2048,4096) row-major -> WxT[g][f] (g<4096,f<1024), WhT[g][f]
__global__ void k_transpose_w(const float* __restrict__ W,
                              __bf16* __restrict__ WxT, __bf16* __restrict__ WhT) {
  size_t idx = (size_t)blockIdx.x * blockDim.x + threadIdx.x;
  if (idx >= (size_t)2048 * G4) return;
  int g  = (int)(idx % G4);
  int f2 = (int)(idx / G4);          // 0..2047 (row of W)
  float v = W[(size_t)f2 * G4 + g];
  if (f2 < FF) WxT[(size_t)g * FF + f2]        = (__bf16)v;
  else         WhT[(size_t)g * HH + (f2 - FF)] = (__bf16)v;
}

// x (N,T,F) fp32 -> Xbf [(t*64+n)][f] bf16
__global__ void k_convert_x(const float* __restrict__ x, __bf16* __restrict__ Xbf) {
  size_t idx = (size_t)blockIdx.x * blockDim.x + threadIdx.x;
  if (idx >= (size_t)NN * TT * FF) return;
  int f = (int)(idx % FF);
  int t = (int)((idx / FF) % TT);
  int n = (int)(idx / ((size_t)FF * TT));
  Xbf[((size_t)t * NN + n) * FF + f] = (__bf16)x[idx];
}

// zero h0 (bf16), c-state (fp32), barrier counter
__global__ void k_init_state(__bf16* __restrict__ h0, float* __restrict__ c,
                             unsigned int* __restrict__ counter) {
  int idx = blockIdx.x * blockDim.x + threadIdx.x;
  if (idx < NN * HH) { h0[idx] = (__bf16)0.0f; c[idx] = 0.0f; }
  if (idx == 0) *counter = 0u;
}

// ---------------- phase 1: big GEMM  Xg = Xbf @ WxT^T + bias ----------------
// one wave -> 16x64 output tile; ping-pong double-buffered A/B fragments.
__global__ __launch_bounds__(256) void
k_gemm_x(const __bf16* __restrict__ Xbf, const __bf16* __restrict__ WxT,
         const float* __restrict__ bias, float* __restrict__ Xg) {
  const int lane = threadIdx.x & 31;
  const int wave = threadIdx.x >> 5;
  const int w = blockIdx.x * 8 + wave;           // 0 .. 2048*64-1
  const int rowTile = w >> 6;                    // 0..2047
  const int colGrp  = w & 63;                    // 64 cols each
  const int ln16 = lane & 15;
  const int aSel = (lane < 16) ? 0 : 16;         // A lane K-offset (contiguous k = 16h + s)
  const int bSel = (lane < 16) ? 0 : 16;         // B lane K-offset (contiguous k = 16h + s)

  const __bf16* aRow = Xbf + ((size_t)rowTile * 16 + ln16) * FF;
  const __bf16* bPtr[4];
  for (int s = 0; s < 4; ++s)
    bPtr[s] = WxT + (size_t)(colGrp * 64 + s * 16 + ln16) * FF + bSel;

  v8f acc[4];
  for (int s = 0; s < 4; ++s) {
    float bv = bias[colGrp * 64 + s * 16 + ln16];
    for (int r = 0; r < 8; ++r) acc[s][r] = bv;  // bias broadcast down rows
  }

  union AV { v16bf v; v8bf h[2]; };
  AV a[2];
  v16bf b[2][4];
  auto loadFrag = [&](int buf, int kb) {
    a[buf].h[0] = *(const v8bf*)(aRow + kb + aSel);
    a[buf].h[1] = *(const v8bf*)(aRow + kb + aSel + 8);
    for (int s = 0; s < 4; ++s) b[buf][s] = *(const v16bf*)(bPtr[s] + kb);
  };

  loadFrag(0, 0);
  for (int kb = 0; kb < FF; kb += 64) {          // FF % 64 == 0
    loadFrag(1, kb + 32);                        // always in range
    for (int s = 0; s < 4; ++s)
      acc[s] = __builtin_amdgcn_wmma_f32_16x16x32_bf16(
          false, a[0].v, false, b[0][s], (short)0, acc[s], false, false);
    if (kb + 64 < FF) loadFrag(0, kb + 64);
    for (int s = 0; s < 4; ++s)
      acc[s] = __builtin_amdgcn_wmma_f32_16x16x32_bf16(
          false, a[1].v, false, b[1][s], (short)0, acc[s], false, false);
  }

  // C/D layout: VGPR r -> row r + (lane<16?0:8), col = lane&15. Stream out NT.
  const int rbase = rowTile * 16 + ((lane < 16) ? 0 : 8);
  for (int s = 0; s < 4; ++s) {
    int col = colGrp * 64 + s * 16 + ln16;
    for (int r = 0; r < 8; ++r)
      __builtin_nontemporal_store(acc[s][r], &Xg[(size_t)(rbase + r) * G4 + col]);
  }
}

// ---------------- phase 2: persistent recurrent kernel ----------------------
// 128 blocks x 8 waves. Wave quad (gates i,f,o,m) shares one (rowTile,colTile)
// pair -> LSTM pointwise update stays block-local (LDS), 1 grid barrier/step.
// Each wave's Wh B-fragments (32KB) are staged once into LDS (async copy) in
// the exact per-lane WMMA layout; hot loop reads B with a single ds access.
__global__ __launch_bounds__(256) void
k_lstm(const float* __restrict__ Xg, const __bf16* __restrict__ WhT,
       __bf16* __restrict__ hbuf /*[2][64*1024]*/, float* __restrict__ cstate,
       float* __restrict__ z, float* __restrict__ hlast, float* __restrict__ clast,
       unsigned int* __restrict__ counter) {
  extern __shared__ char smem[];
  __bf16* Bs = (__bf16*)smem;                              // [8][32][32][16]
  float (*gsm)[4][16][16] = (float (*)[4][16][16])(smem + BS_BYTES);

  const int lane    = threadIdx.x & 31;
  const int wave    = threadIdx.x >> 5;          // 0..7
  const int pairSel = wave >> 2;                 // 0..1
  const int gate    = wave & 3;                  // i,f,o,m
  const int pairId  = blockIdx.x * 2 + pairSel;  // 0..255
  const int rowTile = pairId >> 6;               // 0..3  (batch tiles)
  const int colTile = pairId & 63;               // 0..63 (hidden tiles)
  const int ln16    = lane & 15;
  const int hiHalf  = (lane < 16) ? 0 : 1;
  const int aSel    = hiHalf ? 16 : 0;

  const int aRowIdx = rowTile * 16 + ln16;
  const int bCol    = gate * HH + colTile * 16 + ln16;
  const __bf16* bBase = WhT + (size_t)bCol * HH + (hiHalf ? 16 : 0);

  // ---- one-time async stage of this wave's B fragments into LDS -----------
  {
    const unsigned dbase =
        (unsigned)(size_t)Bs + (unsigned)(((wave * 32) * 32 + lane) * 32);
    for (int kb32 = 0; kb32 < 32; ++kb32) {
      const __bf16* s0 = bBase + kb32 * 32;          // 32 bytes per lane
      const unsigned d0 = dbase + (unsigned)(kb32 * 1024);
      asm volatile("global_load_async_to_lds_b128 %0, %1, off"
                   :: "v"(d0), "v"(s0) : "memory");
      asm volatile("global_load_async_to_lds_b128 %0, %1, off offset:16"
                   :: "v"(d0), "v"(s0) : "memory");
    }
    asm volatile("s_wait_asynccnt 0x0" ::: "memory");
    __syncthreads();
  }
  const v16bf* bFrag = (const v16bf*)Bs + (wave * 32 * 32 + lane); // +32/kb32

  for (int t = 0; t < TT; ++t) {
    const __bf16* hprev = hbuf + (size_t)(t & 1) * (NN * HH);
    __bf16*       hnext = hbuf + (size_t)((t + 1) & 1) * (NN * HH);

    // accumulator = precomputed x-gates (Xg row = t*64 + n)
    v8f acc;
    {
      const float* xg = Xg + ((size_t)t * NN + rowTile * 16 + hiHalf * 8) * G4
                           + gate * HH + colTile * 16 + ln16;
      for (int r = 0; r < 8; ++r) acc[r] = xg[(size_t)r * G4];
    }

    // acc += h_{t-1} @ Wh  (one 16x16 tile; A ping-ponged, B from LDS)
    const __bf16* aRow = hprev + (size_t)aRowIdx * HH;
    union AV { v16bf v; v8bf h[2]; };
    AV a[2];
    auto loadA = [&](int buf, int kb) {
      a[buf].h[0] = *(const v8bf*)(aRow + kb + aSel);
      a[buf].h[1] = *(const v8bf*)(aRow + kb + aSel + 8);
    };
    loadA(0, 0);
    for (int kb32 = 0; kb32 < 32; kb32 += 2) {
      loadA(1, (kb32 + 1) * 32);                 // always in range (kb32 even)
      v16bf b0 = bFrag[kb32 * 32];
      acc = __builtin_amdgcn_wmma_f32_16x16x32_bf16(
          false, a[0].v, false, b0, (short)0, acc, false, false);
      if (kb32 + 2 < 32) loadA(0, (kb32 + 2) * 32);
      v16bf b1 = bFrag[(kb32 + 1) * 32];
      acc = __builtin_amdgcn_wmma_f32_16x16x32_bf16(
          false, a[1].v, false, b1, (short)0, acc, false, false);
    }

    // park gate tile in LDS for the pointwise phase
    for (int r = 0; r < 8; ++r)
      gsm[pairSel][gate][r + hiHalf * 8][ln16] = acc[r];
    __syncthreads();

    // LSTM gating: 512 elements (2 pairs x 256) over 256 threads
    for (int e = threadIdx.x; e < 512; e += 256) {
      int p   = e >> 8;
      int el  = e & 255;
      int m   = el >> 4;
      int j16 = el & 15;
      int pid = blockIdx.x * 2 + p;
      int n = (pid >> 6) * 16 + m;
      int j = (pid & 63) * 16 + j16;
      float iv = gsm[p][0][m][j16];
      float fv = gsm[p][1][m][j16];
      float ov = gsm[p][2][m][j16];
      float mv = gsm[p][3][m][j16];
      float si = 1.0f / (1.0f + __expf(-iv));
      float sf = 1.0f / (1.0f + __expf(-fv));
      float so = 1.0f / (1.0f + __expf(-ov));
      float cold = cstate[n * HH + j];
      float cnew = sf * cold + si * tanhf(mv);
      float hv   = so * tanhf(cnew);
      cstate[n * HH + j] = cnew;
      __builtin_nontemporal_store(hv, &z[((size_t)n * TT + t) * HH + j]);
      hnext[n * HH + j] = (__bf16)hv;
      if (t == TT - 1) { hlast[n * HH + j] = hv; clast[n * HH + j] = cnew; }
    }

    // grid-wide barrier: make h_t visible before anyone starts step t+1
    __threadfence();
    __syncthreads();
    if (threadIdx.x == 0) {
      __hip_atomic_fetch_add(counter, 1u, __ATOMIC_RELEASE, __HIP_MEMORY_SCOPE_AGENT);
      const unsigned target = (unsigned)RNN_BLOCKS * (unsigned)(t + 1);
      while (__hip_atomic_load(counter, __ATOMIC_ACQUIRE, __HIP_MEMORY_SCOPE_AGENT) < target)
        __builtin_amdgcn_s_sleep(1);
    }
    __syncthreads();
  }
}

// ---------------- host launcher ---------------------------------------------

extern "C" void kernel_launch(void* const* d_in, const int* in_sizes, int n_in,
                              void* d_out, int out_size, void* d_ws, size_t ws_size,
                              hipStream_t stream) {
  const float* x    = (const float*)d_in[0];   // (64,512,1024)
  const float* W    = (const float*)d_in[1];   // (2048,4096)
  const float* bias = (const float*)d_in[2];   // (1,4096)

  float* z     = (float*)d_out;                          // (64,512,1024)
  float* hlast = z + (size_t)NN * TT * HH;               // (64,1024)
  float* clast = hlast + (size_t)NN * HH;                // (64,1024)

  // workspace carve-up (256B aligned slabs)
  char* ws = (char*)d_ws;
  size_t off = 0;
  auto take = [&](size_t bytes) { char* p = ws + off; off = (off + bytes + 255) & ~(size_t)255; return p; };
  __bf16* Xbf = (__bf16*)take((size_t)NT * FF * sizeof(__bf16));   // 64 MB
  __bf16* WxT = (__bf16*)take((size_t)G4 * FF * sizeof(__bf16));   //  8 MB
  __bf16* WhT = (__bf16*)take((size_t)G4 * HH * sizeof(__bf16));   //  8 MB
  float*  Xg  = (float*) take((size_t)NT * G4 * sizeof(float));    // 512 MB
  __bf16* hbuf= (__bf16*)take((size_t)2 * NN * HH * sizeof(__bf16));
  float*  cst = (float*) take((size_t)NN * HH * sizeof(float));
  unsigned int* counter = (unsigned int*)take(256);
  (void)ws_size; (void)in_sizes; (void)n_in; (void)out_size;

  // prep
  {
    size_t tot = (size_t)2048 * G4;
    k_transpose_w<<<dim3((unsigned)((tot + 255) / 256)), dim3(256), 0, stream>>>(W, WxT, WhT);
  }
  {
    size_t tot = (size_t)NN * TT * FF;
    k_convert_x<<<dim3((unsigned)((tot + 255) / 256)), dim3(256), 0, stream>>>(x, Xbf);
  }
  k_init_state<<<dim3((NN * HH + 255) / 256), dim3(256), 0, stream>>>(hbuf, cst, counter);

  // phase 1: 2048 row-tiles x 64 col-groups = 131072 waves / 8 per block
  k_gemm_x<<<dim3(16384), dim3(256), 0, stream>>>(Xbf, WxT, bias, Xg);

  // phase 2: persistent recurrent kernel (264KB dynamic LDS per WGP)
  k_lstm<<<dim3(RNN_BLOCKS), dim3(256), LDS_TOTAL, stream>>>(
      Xg, WhT, hbuf, cst, z, hlast, clast, counter);
}
